// Encoder_12635793785031
// MI455X (gfx1250) — matrix operand order repaired and checked by reading, not compiled, not run on hardware
//
#include <hip/hip_runtime.h>
#include <cstdint>
#include <cstddef>

// ---------------------------------------------------------------------------
// CDNA5 (gfx1250) types / helpers
// ---------------------------------------------------------------------------
typedef __attribute__((ext_vector_type(16))) __bf16 v16bf;
typedef __attribute__((ext_vector_type(8)))  float  v8f;
typedef unsigned int v4u_t __attribute__((ext_vector_type(4)));
typedef int v8i_t __attribute__((ext_vector_type(8)));
typedef int v4i_t __attribute__((ext_vector_type(4)));

typedef __attribute__((address_space(1))) v4i_t g_v4i;   // global int4
typedef __attribute__((address_space(3))) v4i_t l_v4i;   // LDS int4
typedef __attribute__((address_space(3))) unsigned short lds_u16;

struct alignas(16) U4 { unsigned v[4]; };
union FragU { v16bf f; U4 q[2]; };

__device__ __forceinline__ unsigned short f2bf(float x) {
  unsigned u = __float_as_uint(x);
  unsigned r = u + 0x7FFFu + ((u >> 16) & 1u);   // round-to-nearest-even
  return (unsigned short)(r >> 16);
}

// A-matrix fragment (16x32 bf16): lane m=lane&15, half=lane>>4.
// elements 0..7  = K in [8*half,   8*half+8)   -> chunk at p
// elements 8..15 = K in [16+8*half,16+8*half+8)-> chunk at p+16
__device__ __forceinline__ v16bf fragA(const unsigned short* p) {
  FragU f;
  f.q[0] = *(const U4*)p;
  f.q[1] = *(const U4*)(p + 16);
  return f.f;
}
// B-matrix fragment (32x16 bf16): lane n=lane&15, half=lane>>4.
// elements 0..7  = K in [8*half, 8*half+8), elements 8..15 = K in [16+8*half,16+8*half+8)
__device__ __forceinline__ v16bf fragB(const unsigned short* p) {
  FragU f;
  f.q[0] = *(const U4*)p;
  f.q[1] = *(const U4*)(p + 16);
  return f.f;
}

__device__ __forceinline__ v8f wmma_bf16(v16bf a, v16bf b, v8f c) {
  return __builtin_amdgcn_wmma_f32_16x16x32_bf16(false, a, false, b, (short)0, c,
                                                 false, false);
}

__device__ __forceinline__ v8f zero8() {
  v8f z;
#pragma unroll
  for (int i = 0; i < 8; i++) z[i] = 0.f;
  return z;
}

// ---- CDNA5 async global->LDS (guarded: falls back to VGPR round-trip) ------
#if __has_builtin(__builtin_amdgcn_global_load_async_to_lds_b128)
#define GEMM_ASYNC 1
__device__ __forceinline__ void async_cp16(const unsigned short* g, unsigned short* l) {
  __builtin_amdgcn_global_load_async_to_lds_b128(
      (g_v4i*)const_cast<unsigned short*>(g), (l_v4i*)l, 0, 0);
}
#else
#define GEMM_ASYNC 0
#endif

__device__ __forceinline__ void wait_async_le3() {
#if __has_builtin(__builtin_amdgcn_s_wait_asynccnt)
  __builtin_amdgcn_s_wait_asynccnt(3);
#else
  asm volatile("s_wait_asynccnt 0x3" ::: "memory");
#endif
}
__device__ __forceinline__ void wait_async_le0() {
#if __has_builtin(__builtin_amdgcn_s_wait_asynccnt)
  __builtin_amdgcn_s_wait_asynccnt(0);
#else
  asm volatile("s_wait_asynccnt 0x0" ::: "memory");
#endif
}

#if __has_builtin(__builtin_amdgcn_tensor_load_to_lds)
#define HAVE_TDM 1
#else
#define HAVE_TDM 0
#endif

#define FLAG_BIAS 1
#define FLAG_ACC  2

// ---------------------------------------------------------------------------
// WMMA GEMM: Y[M,N] (f32) = A[M,Kd] (bf16 row-major) * W[N,Kd]^T (bf16) [+bias][+=Y]
// Rows of A are tokens (b,t); `shift` implements K=3 conv taps with zero pad at
// batch boundaries (rows per batch = Tdim).
// Block tile 128x64, 8 waves (4x2), wave tile 32x32 (2x2 WMMA 16x16x32 tiles).
// Async path: GLOBAL_LOAD_ASYNC_TO_LDS_B128 + double-buffered LDS so stage k+1
// global traffic overlaps stage k WMMAs (ASYNCcnt completes in order; each wave
// issues exactly 3 async loads/stage, so s_wait_asynccnt<=3 fences stage k).
// ---------------------------------------------------------------------------
__global__ __launch_bounds__(256, 2)
void k_gemm(const unsigned short* __restrict__ A, const unsigned short* __restrict__ W,
            float* __restrict__ Y, const float* __restrict__ bias,
            int M, int N, int Kd, int Tdim, int shift, int flags)
{
  __shared__ unsigned short As[2 * 128 * 40];  // 32 bf16 + 8 pad per row (80B rows)
  __shared__ unsigned short Bs[2 * 64 * 40];

  const int tid  = threadIdx.x;
  const int wave = tid >> 5;
  const int lane = tid & 31;
  const int wm = wave >> 1, wn = wave & 1;      // 4(M) x 2(N) waves
  const int m15 = lane & 15, h2 = lane >> 4;
  const int bm = blockIdx.y * 128, bn = blockIdx.x * 64;

  v8f acc[2][2];
#pragma unroll
  for (int i = 0; i < 2; i++)
#pragma unroll
    for (int j = 0; j < 2; j++) acc[i][j] = zero8();

  // Per-thread load assignments
  const int ar = tid >> 1, ap = tid & 1;        // A: 128 rows x 2 halves (16 bf16 each)
  const int br = tid >> 2, bp = tid & 3;        // B: 64 rows x 4 chunks (8 bf16 each)
  const int ag = bm + ar;
  const int at = ag % Tdim;
  const bool aok = (unsigned)(at + shift) < (unsigned)Tdim;
  const unsigned short* aptr = A + (size_t)(ag + shift) * Kd + ap * 16;
  const unsigned short* bptr = W + (size_t)(bn + br) * Kd + bp * 8;
  const int nK = Kd / 32;

#if GEMM_ASYNC
  if (!aok) {                      // conv-tap boundary rows: zero once, never loaded
    U4 z;
#pragma unroll
    for (int i = 0; i < 4; i++) z.v[i] = 0u;
    *(U4*)&As[ar * 40 + ap * 16]           = z;
    *(U4*)&As[ar * 40 + ap * 16 + 8]       = z;
    *(U4*)&As[5120 + ar * 40 + ap * 16]     = z;
    *(U4*)&As[5120 + ar * 40 + ap * 16 + 8] = z;
  }
  if (aok) {
    async_cp16(aptr,     &As[ar * 40 + ap * 16]);
    async_cp16(aptr + 8, &As[ar * 40 + ap * 16 + 8]);
  }
  async_cp16(bptr, &Bs[br * 40 + bp * 8]);

  for (int k = 0; k < nK; k++) {
    const int cur  = k & 1;
    const int aoff = cur * 5120, boff = cur * 2560;
    if (k + 1 < nK) {
      const int noff = (cur ^ 1) * 5120, nboff = (cur ^ 1) * 2560;
      const unsigned short* an = aptr + (size_t)(k + 1) * 32;
      if (aok) {
        async_cp16(an,     &As[noff + ar * 40 + ap * 16]);
        async_cp16(an + 8, &As[noff + ar * 40 + ap * 16 + 8]);
      }
      async_cp16(bptr + (size_t)(k + 1) * 32, &Bs[nboff + br * 40 + bp * 8]);
      wait_async_le3();            // stage k landed; stage k+1 stays in flight
    } else {
      wait_async_le0();
    }
    __syncthreads();

    v16bf af[2], bf[2];
#pragma unroll
    for (int i = 0; i < 2; i++)
      af[i] = fragA(&As[aoff + (wm * 32 + i * 16 + m15) * 40 + 8 * h2]);
#pragma unroll
    for (int j = 0; j < 2; j++)
      bf[j] = fragB(&Bs[boff + (wn * 32 + j * 16 + m15) * 40 + 8 * h2]);

#pragma unroll
    for (int i = 0; i < 2; i++)
#pragma unroll
      for (int j = 0; j < 2; j++)
        acc[i][j] = wmma_bf16(af[i], bf[j], acc[i][j]);

    __syncthreads();
  }
#else
  for (int k0 = 0; k0 < Kd; k0 += 32) {
    U4 a0, a1;
    if (aok) {
      a0 = *(const U4*)(aptr + k0);
      a1 = *(const U4*)(aptr + k0 + 8);
    } else {
#pragma unroll
      for (int i = 0; i < 4; i++) { a0.v[i] = 0u; a1.v[i] = 0u; }
    }
    const U4 b0 = *(const U4*)(bptr + k0);
    *(U4*)&As[ar * 40 + ap * 16]     = a0;
    *(U4*)&As[ar * 40 + ap * 16 + 8] = a1;
    *(U4*)&Bs[br * 40 + bp * 8]      = b0;
    __syncthreads();

    if (k0 + 32 < Kd) {
      __builtin_prefetch(aptr + k0 + 32, 0, 3);
      __builtin_prefetch(bptr + k0 + 32, 0, 3);
    }

    v16bf af[2], bf[2];
#pragma unroll
    for (int i = 0; i < 2; i++)
      af[i] = fragA(&As[(wm * 32 + i * 16 + m15) * 40 + 8 * h2]);
#pragma unroll
    for (int j = 0; j < 2; j++)
      bf[j] = fragB(&Bs[(wn * 32 + j * 16 + m15) * 40 + 8 * h2]);

#pragma unroll
    for (int i = 0; i < 2; i++)
#pragma unroll
      for (int j = 0; j < 2; j++)
        acc[i][j] = wmma_bf16(af[i], bf[j], acc[i][j]);

    __syncthreads();
  }
#endif

  // Epilogue: C/D layout -> element (M = r + 8*half, N = lane&15) in VGPR r
#pragma unroll
  for (int j = 0; j < 2; j++) {
    const int col = bn + wn * 32 + j * 16 + m15;
    const float bv = (flags & FLAG_BIAS) ? bias[col] : 0.f;
#pragma unroll
    for (int i = 0; i < 2; i++) {
#pragma unroll
      for (int r = 0; r < 8; r++) {
        const int row = bm + wm * 32 + i * 16 + r + 8 * h2;
        float* p = &Y[(size_t)row * N + col];
        float v = acc[i][j][r] + bv;
        if (flags & FLAG_ACC) v += *p;
        *p = v;
      }
    }
  }
}

// ---------------------------------------------------------------------------
// Flash attention: one block per (b,h, 64 query rows); K and V^T fully in LDS.
// 128 threads = 4 waves, each wave owns a 16-row query tile.
// K tile (512x64 bf16, row stride C) is DMA'd by the Tensor Data Mover with
// hardware row padding (32-dword interval + 4-dword pad -> LDS stride 72).
// ---------------------------------------------------------------------------
__global__ __launch_bounds__(128, 1)
void k_attn(const unsigned short* __restrict__ qb, const unsigned short* __restrict__ kb,
            const unsigned short* __restrict__ vb, const float* __restrict__ mask,
            unsigned short* __restrict__ ob, int Hn, int Tn, int KCn, float scale)
{
  extern __shared__ unsigned short smem[];
  unsigned short* Ks = smem;                     // [512][72]  (s-major, d contig)
  unsigned short* Vt = smem + 512 * 72;          // [64][520]  (d-major, s contig)
  unsigned short* Ps = Vt + 64 * 520;            // 4 waves x [16][40]

  const int Cn  = Hn * KCn;
  const int bh  = blockIdx.y, b = bh / Hn, h = bh % Hn;
  const int tid = threadIdx.x, wave = tid >> 5, lane = tid & 31;
  const int m15 = lane & 15, h2 = lane >> 4;
  const size_t base = (size_t)b * Tn * Cn + (size_t)h * KCn;

  // Stage V transposed into LDS (manual scatter; TDM cannot transpose)
  for (int idx = tid; idx < Tn * 8; idx += 128) {
    const int row = idx >> 3, seg = idx & 7;
    const U4 e = *(const U4*)(vb + base + (size_t)row * Cn + seg * 8);
    const unsigned short* ev = (const unsigned short*)&e;
#pragma unroll
    for (int j = 0; j < 8; j++) Vt[(seg * 8 + j) * 520 + row] = ev[j];
  }

#if HAVE_TDM
  if (wave == 0) {                 // whole wave takes branch: EXEC all ones
    const unsigned long long ga = (unsigned long long)(size_t)(kb + base);
    const unsigned lds_off = (unsigned)(size_t)(lds_u16*)Ks;
    // D# group0: count=1, lds_addr, 57b global addr, type=2
    const v4u_t g0 = {1u, lds_off, (unsigned)(ga & 0xFFFFFFFFu),
                      (unsigned)((ga >> 32) & 0x1FFFFFFu) | (2u << 30)};
    // D# group1: data_size=2B, pad_enable, pad_interval=32dw, pad_amount=4dw,
    //            tensor_dim0=64, tensor_dim1=512, tile 64x512, dim0_stride=Cn
    const v8i_t g1 = {(int)((1u << 16) | (1u << 20) | (4u << 22) | (3u << 25)),
                      (int)(64u << 16),        // tensor_dim0[15:0] @ bits63:48
                      (int)(512u << 16),       // tensor_dim1[15:0] @ bits95:80
                      (int)(64u << 16),        // tile_dim0 @ bits127:112
                      (int)512,                // tile_dim1 @ bits143:128
                      (int)(unsigned)Cn,       // tensor_dim0_stride[31:0]
                      0, 0};
    const v4i_t g2 = {0, 0, 0, 0};
    const v4i_t g3 = {0, 0, 0, 0};
    const v8i_t g4 = {0, 0, 0, 0, 0, 0, 0, 0};
    __builtin_amdgcn_tensor_load_to_lds(g0, g1, g2, g3, g4, 0);
#if __has_builtin(__builtin_amdgcn_s_wait_tensorcnt)
    __builtin_amdgcn_s_wait_tensorcnt(0);
#else
    asm volatile("s_wait_tensorcnt 0x0" ::: "memory");
#endif
  }
#else
  for (int idx = tid; idx < Tn * 8; idx += 128) {
    const int row = idx >> 3, seg = idx & 7;
    const U4 d = *(const U4*)(kb + base + (size_t)row * Cn + seg * 8);
    *(U4*)&Ks[row * 72 + seg * 8] = d;
  }
#endif
  __syncthreads();

  const int t0 = blockIdx.x * 64 + wave * 16;

  v16bf qf[2];
#pragma unroll
  for (int ks = 0; ks < 2; ks++)
    qf[ks] = fragA(qb + base + (size_t)(t0 + m15) * Cn + ks * 32 + 8 * h2);

  float qm[8];
#pragma unroll
  for (int r = 0; r < 8; r++) qm[r] = mask[b * Tn + t0 + r + 8 * h2];

  float mrun[8], lrun[8];
  v8f O[4];
#pragma unroll
  for (int r = 0; r < 8; r++) { mrun[r] = -1e30f; lrun[r] = 0.f; }
#pragma unroll
  for (int nt = 0; nt < 4; nt++) O[nt] = zero8();

  unsigned short* P = Ps + wave * 16 * 40;

  for (int s0 = 0; s0 < Tn; s0 += 32) {
    v8f sc0 = zero8(), sc1 = zero8();
#pragma unroll
    for (int ks = 0; ks < 2; ks++) {
      sc0 = wmma_bf16(qf[ks], fragB(&Ks[(s0 + m15) * 72 + ks * 32 + 8 * h2]), sc0);
      sc1 = wmma_bf16(qf[ks], fragB(&Ks[(s0 + 16 + m15) * 72 + ks * 32 + 8 * h2]), sc1);
    }
    const float km0 = mask[b * Tn + s0 + m15];
    const float km1 = mask[b * Tn + s0 + 16 + m15];

    float corr[8];
#pragma unroll
    for (int r = 0; r < 8; r++) {
      float v0 = sc0[r] * scale; if (qm[r] * km0 == 0.f) v0 = -1e4f;
      float v1 = sc1[r] * scale; if (qm[r] * km1 == 0.f) v1 = -1e4f;
      float mx = fmaxf(v0, v1);
#pragma unroll
      for (int off = 8; off >= 1; off >>= 1) mx = fmaxf(mx, __shfl_xor(mx, off, 32));
      const float mnew = fmaxf(mrun[r], mx);
      const float cr = __expf(mrun[r] - mnew);
      const float p0 = __expf(v0 - mnew);
      const float p1 = __expf(v1 - mnew);
      float rs = p0 + p1;
#pragma unroll
      for (int off = 8; off >= 1; off >>= 1) rs += __shfl_xor(rs, off, 32);
      mrun[r] = mnew;
      lrun[r] = lrun[r] * cr + rs;
      corr[r] = cr;
      // stage P (bf16) in per-wave LDS so it can re-enter WMMA A-layout
      P[(r + 8 * h2) * 40 + m15]      = f2bf(p0);
      P[(r + 8 * h2) * 40 + 16 + m15] = f2bf(p1);
    }
#pragma unroll
    for (int nt = 0; nt < 4; nt++)
#pragma unroll
      for (int r = 0; r < 8; r++) O[nt][r] *= corr[r];

    const v16bf pa = fragA(&P[m15 * 40 + 8 * h2]);   // same-wave LDS: DS in-order
#pragma unroll
    for (int nt = 0; nt < 4; nt++)
      O[nt] = wmma_bf16(pa, fragB(&Vt[(nt * 16 + m15) * 520 + s0 + 8 * h2]), O[nt]);
  }

#pragma unroll
  for (int r = 0; r < 8; r++) {
    const int row = t0 + r + 8 * h2;
    const float inv = 1.f / lrun[r];
#pragma unroll
    for (int nt = 0; nt < 4; nt++)
      ob[base + (size_t)row * Cn + nt * 16 + m15] = f2bf(O[nt][r] * inv);
  }
}

// ---------------------------------------------------------------------------
// Elementwise / reduction kernels
// ---------------------------------------------------------------------------
__global__ void k_cvt(const float* __restrict__ in, unsigned short* __restrict__ out, size_t n) {
  const size_t i = (size_t)blockIdx.x * 256 + threadIdx.x;
  if (i < n) out[i] = f2bf(in[i]);
}

__global__ void k_maskcvt(float* __restrict__ x, unsigned short* __restrict__ xb,
                          const float* __restrict__ mask, int Cn, size_t n, int mutate) {
  const size_t i = (size_t)blockIdx.x * 256 + threadIdx.x;
  if (i >= n) return;
  const float m = mask[i / Cn];
  const float v = x[i] * m;
  if (mutate) x[i] = v;
  xb[i] = f2bf(v);
}

__global__ void k_relumaskcvt(const float* __restrict__ in, unsigned short* __restrict__ out,
                              const float* __restrict__ mask, int Fn, size_t n) {
  const size_t i = (size_t)blockIdx.x * 256 + threadIdx.x;
  if (i >= n) return;
  float v = in[i];
  v = v > 0.f ? v : 0.f;
  out[i] = f2bf(v * mask[i / Fn]);
}

// W[o][c][k] (f32) -> dst[k][o][c] (bf16): one contiguous [O x Cc] matrix per tap
__global__ void k_tap3(const float* __restrict__ src, unsigned short* __restrict__ dst,
                       int O, int Cc) {
  const size_t n = (size_t)O * Cc * 3;
  const size_t i = (size_t)blockIdx.x * 256 + threadIdx.x;
  if (i >= n) return;
  const int k = (int)(i % 3);
  const size_t oc = i / 3;
  dst[(size_t)k * O * Cc + oc] = f2bf(src[i]);
}

// x_in[b][c][t] -> x32[b][t][c]
__global__ void k_tin(const float* __restrict__ xin, float* __restrict__ x32,
                      int Cn, int Tn, size_t n) {
  const size_t i = (size_t)blockIdx.x * 256 + threadIdx.x;
  if (i >= n) return;
  const int c = (int)(i % Cn);
  const size_t bt = i / Cn;
  const int t = (int)(bt % Tn);
  const size_t b = bt / Tn;
  x32[i] = xin[(b * Cn + c) * Tn + t];
}

// d_out[b][c][t] = x32[b][t][c] * mask[b][t]
__global__ void k_tout(const float* __restrict__ x32, const float* __restrict__ mask,
                       float* __restrict__ out, int Cn, int Tn, size_t n) {
  const size_t i = (size_t)blockIdx.x * 256 + threadIdx.x;
  if (i >= n) return;
  const int t = (int)(i % Tn);
  const size_t bc = i / Tn;
  const int c = (int)(bc % Cn);
  const size_t b = bc / Cn;
  out[i] = x32[(b * Tn + t) * Cn + c] * mask[b * Tn + t];
}

// RoPE on first 32 dims of each 64-dim head: pair (j, j+16), freq 10000^{-j/16}
__global__ void k_rope(float* __restrict__ q, int Tn, int Hn, int KCn, size_t total) {
  const size_t i = (size_t)blockIdx.x * 256 + threadIdx.x;
  if (i >= total) return;
  const int j = (int)(i & 15);
  const size_t r = i >> 4;
  const int hh = (int)(r % Hn);
  const size_t bt = r / Hn;
  const int t = (int)(bt % Tn);
  const float theta = __powf(10000.f, -(float)j / 16.f);
  float sa, ca;
  __sincosf((float)t * theta, &sa, &ca);
  float* p = q + bt * (size_t)(Hn * KCn) + hh * KCn;
  const float x0 = p[j], x1 = p[j + 16];
  p[j]      = x0 * ca - x1 * sa;
  p[j + 16] = x1 * ca + x0 * sa;
}

// Reference-faithful LayerNorm quirk: var = Var((x-mean)^2, ddof=1)
// x <- (x + y[*mask]) normalized * gamma + beta       (in place, C=768)
__global__ __launch_bounds__(256)
void k_ln(float* __restrict__ x, const float* __restrict__ y,
          const float* __restrict__ g, const float* __restrict__ be,
          const float* __restrict__ mask, int useMask)
{
  __shared__ float red[256];
  const int Cc = 768;
  const int tok = blockIdx.x, tid = threadIdx.x;
  const float mv = useMask ? mask[tok] : 1.f;
  const size_t bbase = (size_t)tok * Cc;

  const float v0 = x[bbase + tid]       + y[bbase + tid]       * mv;
  const float v1 = x[bbase + tid + 256] + y[bbase + tid + 256] * mv;
  const float v2 = x[bbase + tid + 512] + y[bbase + tid + 512] * mv;

  red[tid] = v0 + v1 + v2;
  __syncthreads();
  for (int o2 = 128; o2 > 0; o2 >>= 1) { if (tid < o2) red[tid] += red[tid + o2]; __syncthreads(); }
  const float mean = red[0] / Cc;
  __syncthreads();

  const float d0 = v0 - mean, d1 = v1 - mean, d2 = v2 - mean;
  const float q0 = d0 * d0, q1 = d1 * d1, q2 = d2 * d2;

  red[tid] = q0 + q1 + q2;
  __syncthreads();
  for (int o2 = 128; o2 > 0; o2 >>= 1) { if (tid < o2) red[tid] += red[tid + o2]; __syncthreads(); }
  const float S2 = red[0];
  __syncthreads();

  red[tid] = q0 * q0 + q1 * q1 + q2 * q2;
  __syncthreads();
  for (int o2 = 128; o2 > 0; o2 >>= 1) { if (tid < o2) red[tid] += red[tid + o2]; __syncthreads(); }
  const float S4 = red[0];

  const float msq = S2 / Cc;
  const float var = (S4 - Cc * msq * msq) / (Cc - 1);
  const float inv = rsqrtf(var + 1e-4f);

  x[bbase + tid]       = d0 * inv * g[tid]       + be[tid];
  x[bbase + tid + 256] = d1 * inv * g[tid + 256] + be[tid + 256];
  x[bbase + tid + 512] = d2 * inv * g[tid + 512] + be[tid + 512];
}

// ---------------------------------------------------------------------------
// Launcher
// ---------------------------------------------------------------------------
extern "C" void kernel_launch(void* const* d_in, const int* in_sizes, int n_in,
                              void* d_out, int out_size, void* d_ws, size_t ws_size,
                              hipStream_t stream)
{
  (void)in_sizes; (void)n_in; (void)out_size; (void)ws_size;

  const int Bn = 8, Tn = 512, Cn = 768, Hn = 12, Fn = 3072, Ln = 6, KK = 3, KCn = 64;
  const int Mn = Bn * Tn;  // 4096 tokens

  const float* x_in = (const float*)d_in[0];
  const float* mask = (const float*)d_in[1];
  const float* Wq = (const float*)d_in[2];
  const float* bq = (const float*)d_in[3];
  const float* Wk = (const float*)d_in[4];
  const float* bk = (const float*)d_in[5];
  const float* Wv = (const float*)d_in[6];
  const float* bv = (const float*)d_in[7];
  const float* Wo = (const float*)d_in[8];
  const float* bo = (const float*)d_in[9];
  const float* g1 = (const float*)d_in[10];
  const float* be1 = (const float*)d_in[11];
  const float* Wf1 = (const float*)d_in[12];
  const float* bf1 = (const float*)d_in[13];
  const float* Wf2 = (const float*)d_in[14];
  const float* bf2 = (const float*)d_in[15];
  const float* g2 = (const float*)d_in[16];
  const float* be2 = (const float*)d_in[17];

  char* ws = (char*)d_ws;
  size_t off = 0;
  auto alloc = [&](size_t bytes) -> char* {
    char* p = ws + off;
    off += (bytes + 255) & ~(size_t)255;
    return p;
  };

  const size_t MC4 = (size_t)Mn * Cn * 4, MC2 = (size_t)Mn * Cn * 2;
  const size_t MF4 = (size_t)Mn * Fn * 4, MF2 = (size_t)Mn * Fn * 2;

  float* x32 = (float*)alloc(MC4);          // residual stream [B*T, C]
  float* y32 = (float*)alloc(MC4);          // branch output
  unsigned short* xb = (unsigned short*)alloc(MC2);

  char* sA = alloc(MF4);                    // q32/k32/v32  |  h32 (overlapped)
  float* q32 = (float*)sA;
  float* k32 = (float*)(sA + MC4);
  float* v32 = (float*)(sA + 2 * MC4);
  float* h32 = (float*)sA;

  char* sB = alloc(4 * MC2);                // qb/kb/vb/attn | hb (overlapped, == MF2)
  unsigned short* qb2 = (unsigned short*)sB;
  unsigned short* kb2 = (unsigned short*)(sB + MC2);
  unsigned short* vb2 = (unsigned short*)(sB + 2 * MC2);
  unsigned short* ab  = (unsigned short*)(sB + 3 * MC2);
  unsigned short* hb  = (unsigned short*)sB;
  (void)MF2;

  const size_t CC = (size_t)Cn * Cn;
  unsigned short* wqb = (unsigned short*)alloc(CC * 2);
  unsigned short* wkb = (unsigned short*)alloc(CC * 2);
  unsigned short* wvb = (unsigned short*)alloc(CC * 2);
  unsigned short* wob = (unsigned short*)alloc(CC * 2);
  unsigned short* w1t = (unsigned short*)alloc((size_t)3 * Fn * Cn * 2);
  unsigned short* w2t = (unsigned short*)alloc((size_t)3 * Cn * Fn * 2);

  auto g256 = [](size_t n) { return dim3((unsigned)((n + 255) / 256)); };
  const dim3 blk(256);
  const dim3 gemmC(Cn / 64, Mn / 128);      // N=768
  const dim3 gemmF(Fn / 64, Mn / 128);      // N=3072
  const size_t nMC = (size_t)Mn * Cn, nMF = (size_t)Mn * Fn;
  const size_t attn_lds = (size_t)(512 * 72 + 64 * 520 + 4 * 16 * 40) * 2;

  k_tin<<<g256(nMC), blk, 0, stream>>>(x_in, x32, Cn, Tn, nMC);

  for (int l = 0; l < Ln; l++) {
    const size_t wcc = (size_t)l * CC;
    k_cvt<<<g256(CC), blk, 0, stream>>>(Wq + wcc, wqb, CC);
    k_cvt<<<g256(CC), blk, 0, stream>>>(Wk + wcc, wkb, CC);
    k_cvt<<<g256(CC), blk, 0, stream>>>(Wv + wcc, wvb, CC);
    k_cvt<<<g256(CC), blk, 0, stream>>>(Wo + wcc, wob, CC);
    k_tap3<<<g256((size_t)Fn * Cn * 3), blk, 0, stream>>>(Wf1 + (size_t)l * Fn * Cn * KK, w1t, Fn, Cn);
    k_tap3<<<g256((size_t)Fn * Cn * 3), blk, 0, stream>>>(Wf2 + (size_t)l * Cn * Fn * KK, w2t, Cn, Fn);

    // x = x * mask ; xb = bf16(x)
    k_maskcvt<<<g256(nMC), blk, 0, stream>>>(x32, xb, mask, Cn, nMC, 1);

    // Q, K, V projections
    k_gemm<<<gemmC, blk, 0, stream>>>(xb, wqb, q32, bq + l * Cn, Mn, Cn, Cn, Tn, 0, FLAG_BIAS);
    k_gemm<<<gemmC, blk, 0, stream>>>(xb, wkb, k32, bk + l * Cn, Mn, Cn, Cn, Tn, 0, FLAG_BIAS);
    k_gemm<<<gemmC, blk, 0, stream>>>(xb, wvb, v32, bv + l * Cn, Mn, Cn, Cn, Tn, 0, FLAG_BIAS);

    const size_t nrope = (size_t)Mn * Hn * 16;
    k_rope<<<g256(nrope), blk, 0, stream>>>(q32, Tn, Hn, KCn, nrope);
    k_rope<<<g256(nrope), blk, 0, stream>>>(k32, Tn, Hn, KCn, nrope);

    k_cvt<<<g256(nMC), blk, 0, stream>>>(q32, qb2, nMC);
    k_cvt<<<g256(nMC), blk, 0, stream>>>(k32, kb2, nMC);
    k_cvt<<<g256(nMC), blk, 0, stream>>>(v32, vb2, nMC);

    // Attention (flash, K via TDM, K/V^T resident in WGP LDS)
    const dim3 agrid(Tn / 64, Bn * Hn);
    k_attn<<<agrid, dim3(128), attn_lds, stream>>>(qb2, kb2, vb2, mask, ab, Hn, Tn, KCn, 0.125f);

    // Output projection + residual + LN1
    k_gemm<<<gemmC, blk, 0, stream>>>(ab, wob, y32, bo + l * Cn, Mn, Cn, Cn, Tn, 0, FLAG_BIAS);
    k_ln<<<dim3(Mn), blk, 0, stream>>>(x32, y32, g1 + l * Cn, be1 + l * Cn, mask, 0);

    // FFN: conv1d(K=3) as 3 shifted GEMM taps
    k_maskcvt<<<g256(nMC), blk, 0, stream>>>(x32, xb, mask, Cn, nMC, 0);
    for (int dt = 0; dt < 3; dt++)
      k_gemm<<<gemmF, blk, 0, stream>>>(xb, w1t + (size_t)dt * Fn * Cn, h32, bf1 + l * Fn,
                                        Mn, Fn, Cn, Tn, dt - 1, dt == 0 ? FLAG_BIAS : FLAG_ACC);
    k_relumaskcvt<<<g256(nMF), blk, 0, stream>>>(h32, hb, mask, Fn, nMF);
    for (int dt = 0; dt < 3; dt++)
      k_gemm<<<gemmC, blk, 0, stream>>>(hb, w2t + (size_t)dt * Cn * Fn, y32, bf2 + l * Cn,
                                        Mn, Cn, Fn, Tn, dt - 1, dt == 0 ? FLAG_BIAS : FLAG_ACC);
    k_ln<<<dim3(Mn), blk, 0, stream>>>(x32, y32, g2 + l * Cn, be2 + l * Cn, mask, 1);
  }

  const size_t nout = (size_t)Bn * Cn * Tn;
  k_tout<<<g256(nout), blk, 0, stream>>>(x32, mask, (float*)d_out, Cn, Tn, nout);
}
